// TrainableMemory_88699664597646
// MI455X (gfx1250) — compile-verified, matrix operand order checked by reading, not run on hardware
//
#include <hip/hip_runtime.h>
#include <hip/hip_bf16.h>
#include <cstdint>

typedef _Float16 h16;
typedef __attribute__((ext_vector_type(16))) _Float16 v16h;
typedef __attribute__((ext_vector_type(8)))  float    v8f;
typedef __attribute__((ext_vector_type(4)))  uint32_t vu4;
typedef __attribute__((ext_vector_type(8)))  int      vi8;
typedef __attribute__((ext_vector_type(4)))  int      vi4;

#define NPIX   32768      // 8*64*64
#define CDIM   256
#define MMEM   2024
#define MPAD   2048
#define NCHUNK (MPAD / 32)
#define CHID   64
#define KSTRIDE 272       // padded LDS row stride (halves): 256 data + 16 pad
#define KBYTES  (32 * KSTRIDE)   // halves per K buffer
#define INV_TEMP (1.0f/0.03f)
#define NEG_BIG  (-1.0e30f)

// ---------------------------------------------------------------------------
// Kernel 1: normalize fg/bg memory rows -> f16, zero-pad to MPAD rows.
// grid (MPAD, 2), block 256.
// ---------------------------------------------------------------------------
__global__ void tm_prep_mem(const float* __restrict__ fg,
                            const float* __restrict__ bg,
                            h16* __restrict__ fgn16,
                            h16* __restrict__ bgn16) {
  const int row  = blockIdx.x;
  const int t    = threadIdx.x;           // == channel
  const int lane = t & 31, wave = t >> 5;
  const float* src = (blockIdx.y == 0) ? fg : bg;
  h16*         dst = (blockIdx.y == 0) ? fgn16 : bgn16;

  __shared__ float part[8];
  __shared__ float s_inv;

  float v  = (row < MMEM) ? src[(size_t)row * CDIM + t] : 0.0f;
  float ss = v * v;
  #pragma unroll
  for (int off = 16; off >= 1; off >>= 1) ss += __shfl_xor(ss, off, 32);
  if (lane == 0) part[wave] = ss;
  __syncthreads();
  if (t == 0) {
    float tot = 0.0f;
    #pragma unroll
    for (int w = 0; w < 8; ++w) tot += part[w];
    s_inv = 1.0f / fmaxf(sqrtf(tot), 1e-12f);
  }
  __syncthreads();
  dst[(size_t)row * CDIM + t] = (h16)(v * s_inv);
}

// ---------------------------------------------------------------------------
// Kernel 2: transpose-gather feats, L2-normalize per pixel -> xn16[N,C] (f16),
// plus gate MLPs g_fg[N], g_bg[N].  grid NPIX/32, block 256 (8 waves).
// ---------------------------------------------------------------------------
__global__ void tm_prep_x(const float* __restrict__ feats,
                          const float* __restrict__ w1f, const float* __restrict__ b1f,
                          const float* __restrict__ w2f, const float* __restrict__ b2f,
                          const float* __restrict__ w1b, const float* __restrict__ b1b,
                          const float* __restrict__ w2b, const float* __restrict__ b2b,
                          h16*   __restrict__ xn16,
                          float* __restrict__ gfg,
                          float* __restrict__ gbg) {
  const int t    = threadIdx.x;
  const int lane = t & 31, wave = t >> 5;
  const int n0   = blockIdx.x * 32;
  const int b    = n0 >> 12;              // 4096 hw positions per batch
  const int hw0  = n0 & 4095;

  __shared__ float xs[CDIM][33];          // [channel][pixel], padded
  __shared__ float part[8][32];
  __shared__ float invs[32];

  // Phase A: coalesced gather (lane = pixel) + per-pixel sumsq partials
  float ss = 0.0f;
  #pragma unroll 4
  for (int cc = 0; cc < 32; ++cc) {
    const int c = wave * 32 + cc;
    float v = feats[(((size_t)(b * CDIM + c)) << 12) + hw0 + lane];
    xs[c][lane] = v;
    ss += v * v;
  }
  part[wave][lane] = ss;
  __syncthreads();
  if (t < 32) {
    float tot = 0.0f;
    #pragma unroll
    for (int w = 0; w < 8; ++w) tot += part[w][t];
    invs[t] = 1.0f / fmaxf(sqrtf(tot), 1e-12f);
  }
  __syncthreads();

  // Phase B: write xn16 row-major; thread t -> pixel p, 32 channels
  {
    const int p  = t >> 3;
    const int c0 = (t & 7) * 32;
    const float inv = invs[p];
    v16h o0, o1;
    #pragma unroll
    for (int i = 0; i < 16; ++i) {
      o0[i] = (h16)(xs[c0 + i][p] * inv);
      o1[i] = (h16)(xs[c0 + 16 + i][p] * inv);
    }
    h16* dst = xn16 + (size_t)(n0 + p) * CDIM + c0;
    *(v16h*)dst        = o0;
    *(v16h*)(dst + 16) = o1;
  }

  // Phase C: gate MLPs. thread t -> pixel p = t>>3, hidden slice j0 = (t&7)*8
  {
    const int p  = t >> 3;
    const int j0 = (t & 7) * 8;
    const float inv = invs[p];
    for (int e = 0; e < 2; ++e) {
      const float* W1 = e ? w1b : w1f;
      const float* B1 = e ? b1b : b1f;
      const float* W2 = e ? w2b : w2f;
      const float* B2 = e ? b2b : b2f;
      float hreg[8];
      #pragma unroll
      for (int j = 0; j < 8; ++j) hreg[j] = 0.0f;
      for (int c = 0; c < CDIM; ++c) {
        float xv = xs[c][p] * inv;
        #pragma unroll
        for (int j = 0; j < 8; ++j) hreg[j] += xv * W1[c * CHID + j0 + j];
      }
      float z = 0.0f;
      #pragma unroll
      for (int j = 0; j < 8; ++j) {
        float hv = fmaxf(hreg[j] + B1[j0 + j], 0.0f);
        z += hv * W2[j0 + j];
      }
      #pragma unroll
      for (int off = 4; off >= 1; off >>= 1) z += __shfl_xor(z, off, 8);
      if ((t & 7) == 0) {
        float g = 1.0f / (1.0f + __expf(-(z + B2[0])));
        (e ? gbg : gfg)[n0 + p] = g;
      }
    }
  }
}

// ---------------------------------------------------------------------------
// Issue one TDM load: 32x256 f16 tile (chunkIdx*32 rows) -> LDS, padded rows.
// D# group0: count=1 | lds_addr | global_addr(57b) | type=2
// D# group1: data_size=2B, pad_enable, pad_interval=128dw, pad_amount=8dw,
//            tensor_dim0=256, tensor_dim1=MPAD, tile=256x32, dim0_stride=256
// ---------------------------------------------------------------------------
__device__ __forceinline__ void tm_issue_tdm(const h16* memn16, int chunkIdx,
                                             h16* ldsDst) {
  uint64_t ga = (uint64_t)(uintptr_t)(memn16 + (size_t)chunkIdx * 32 * CDIM);
  uint32_t la = (uint32_t)(uintptr_t)ldsDst;   // low 32 bits = LDS byte offset
  vu4 g0;
  g0[0] = 1u;                                  // count=1, user mode
  g0[1] = la;                                  // lds_addr  [63:32]
  g0[2] = (uint32_t)ga;                        // global_addr low
  g0[3] = (uint32_t)((ga >> 32) & 0x01FFFFFFu) | (2u << 30);  // addr hi | type=2
  vi8 g1;
  g1[0] = (int)((1u << 16) | (1u << 20) | (6u << 22) | (7u << 25));
  g1[1] = (int)(256u << 16);                   // tensor_dim0[15:0] @ bits 63:48
  g1[2] = (int)((uint32_t)MPAD << 16);         // tensor_dim1[15:0] @ bits 95:80
  g1[3] = (int)(256u << 16);                   // tile_dim0 @ bits 127:112
  g1[4] = 32;                                  // tile_dim1
  g1[5] = 256;                                 // tensor_dim0_stride low
  g1[6] = 0;
  g1[7] = 0;
  vi4 z4 = {0, 0, 0, 0};
  vi8 z8 = {0, 0, 0, 0, 0, 0, 0, 0};
  __builtin_amdgcn_tensor_load_to_lds(g0, g1, z4, z4, z8, 0);
}

// ---------------------------------------------------------------------------
// Kernel 3: streaming-softmax attention over memory (V == K).
// Block: 256 thr = 8 waves; wave -> 16 rows x full C.  grid NPIX/128.
// K chunks staged by the Tensor Data Mover, double-buffered.
// pass 0: memfg[n,c] = g_fg * (softmax(QK^T)K)      (write workspace)
// pass 1: out = NCHW( xn + memfg - g_bg * (softmax..K) )
// ---------------------------------------------------------------------------
__global__ void __launch_bounds__(256, 1)
tm_attn(const h16* __restrict__ xn16,
        const h16* __restrict__ memn16,     // [MPAD, C] f16, zero-padded
        const float* __restrict__ gate,     // [N]
        float* __restrict__ memfg,          // [N, C] f32 workspace
        float* __restrict__ out,            // [B,C,H,W] f32 (pass 1 only)
        int pass) {
  const int t    = threadIdx.x;
  const int lane = t & 31, wave = t >> 5;
  const int l15  = lane & 15, half = lane >> 4;
  const int rowbase = blockIdx.x * 128 + wave * 16;

  __shared__ h16 Kbuf [2 * KBYTES];   // double-buffered TDM target, [mrow][c]
  __shared__ h16 KbufT[256 * 32];     // [channel][mrow], stride 32 (64B rows)
  __shared__ h16 Pbuf [8 * 16 * 32];  // per-wave 16x32 P staging
  h16* Pw = Pbuf + wave * (16 * 32);

  const v8f zero8 = {0.f,0.f,0.f,0.f,0.f,0.f,0.f,0.f};
  v8f acc[16];
  #pragma unroll
  for (int i = 0; i < 16; ++i) acc[i] = zero8;
  float mrun[8], lrun[8];
  #pragma unroll
  for (int r = 0; r < 8; ++r) { mrun[r] = NEG_BIG; lrun[r] = 0.0f; }

  const int ldrow = t >> 3;          // KbufT builder: 32 rows
  const int ldc0  = (t & 7) * 32;    // 8 x 32-channel slabs

  if (wave == 0) tm_issue_tdm(memn16, 0, Kbuf);   // prologue: chunk 0 in flight

  for (int chunk = 0; chunk < NCHUNK; ++chunk) {
    __syncthreads();   // all waves done with Kbuf[(chunk+1)&1] and KbufT
    if (wave == 0) {
      if (chunk + 1 < NCHUNK) {
        tm_issue_tdm(memn16, chunk + 1, Kbuf + ((chunk + 1) & 1) * KBYTES);
        __builtin_amdgcn_s_wait_tensorcnt(1);   // in-order: chunk's load done
      } else {
        __builtin_amdgcn_s_wait_tensorcnt(0);
      }
    }
    __syncthreads();   // TDM(chunk) data visible to all waves

    const h16* Kc = Kbuf + (chunk & 1) * KBYTES;

    // build channel-major copy for the P.V B-fragments
    {
      v16h x0 = *(const v16h*)&Kc[ldrow * KSTRIDE + ldc0];
      v16h x1 = *(const v16h*)&Kc[ldrow * KSTRIDE + ldc0 + 16];
      #pragma unroll
      for (int i = 0; i < 16; ++i) {
        KbufT[(ldc0 + i)      * 32 + ldrow] = x0[i];
        KbufT[(ldc0 + 16 + i) * 32 + ldrow] = x1[i];
      }
    }
    __syncthreads();

    // ---- S = (Q . K^T) * INV_TEMP : two 16x16 tiles, K-dim 256 ----
    const int base = chunk * 32;
    v8f s0 = zero8, s1 = zero8;
    #pragma unroll
    for (int kc = 0; kc < 8; ++kc) {
      const int cbase = kc * 32;
      v16h aq = *(const v16h*)(xn16 + (size_t)(rowbase + l15) * CDIM + cbase + half * 16);
      v16h b0 = *(const v16h*)(&Kc[(l15)      * KSTRIDE + cbase + half * 16]);
      v16h b1 = *(const v16h*)(&Kc[(16 + l15) * KSTRIDE + cbase + half * 16]);
      s0 = __builtin_amdgcn_wmma_f32_16x16x32_f16(false, aq, false, b0, (short)0, s0, false, false);
      s1 = __builtin_amdgcn_wmma_f32_16x16x32_f16(false, aq, false, b1, (short)0, s1, false, false);
    }
    const bool oob0 = (base + l15)      >= MMEM;
    const bool oob1 = (base + 16 + l15) >= MMEM;
    #pragma unroll
    for (int r = 0; r < 8; ++r) {
      s0[r] = oob0 ? NEG_BIG : s0[r] * INV_TEMP;
      s1[r] = oob1 ? NEG_BIG : s1[r] * INV_TEMP;
    }

    // ---- online softmax over this 32-col chunk ----
    float rmax[8], sc[8], rsum[8];
    v8f p0, p1;
    #pragma unroll
    for (int r = 0; r < 8; ++r) rmax[r] = fmaxf(s0[r], s1[r]);
    #pragma unroll
    for (int off = 8; off >= 1; off >>= 1)
      #pragma unroll
      for (int r = 0; r < 8; ++r)
        rmax[r] = fmaxf(rmax[r], __shfl_xor(rmax[r], off, 32));
    #pragma unroll
    for (int r = 0; r < 8; ++r) {
      float nm = fmaxf(mrun[r], rmax[r]);
      sc[r]   = __expf(mrun[r] - nm);
      mrun[r] = nm;
      p0[r] = __expf(s0[r] - nm);
      p1[r] = __expf(s1[r] - nm);
      rsum[r] = p0[r] + p1[r];
    }
    #pragma unroll
    for (int off = 8; off >= 1; off >>= 1)
      #pragma unroll
      for (int r = 0; r < 8; ++r)
        rsum[r] += __shfl_xor(rsum[r], off, 32);
    #pragma unroll
    for (int r = 0; r < 8; ++r) lrun[r] = lrun[r] * sc[r] + rsum[r];
    #pragma unroll
    for (int i = 0; i < 16; ++i)
      #pragma unroll
      for (int r = 0; r < 8; ++r) acc[i][r] *= sc[r];

    // ---- P (D layout) -> LDS -> A-fragment layout ----
    #pragma unroll
    for (int r = 0; r < 8; ++r) {
      const int prow = r + 8 * half;
      Pw[prow * 32 + l15]      = (h16)p0[r];
      Pw[prow * 32 + 16 + l15] = (h16)p1[r];
    }
    v16h pa = *(const v16h*)(Pw + l15 * 32 + half * 16);

    // ---- O += P . V  (V == K, channel-major layout) ----
    #pragma unroll
    for (int i = 0; i < 16; ++i) {
      v16h bv = *(const v16h*)(&KbufT[(i * 16 + l15) * 32 + half * 16]);
      acc[i] = __builtin_amdgcn_wmma_f32_16x16x32_f16(false, pa, false, bv, (short)0, acc[i], false, false);
    }
  }

  // ---- epilogue ----
  float invl[8], grow[8];
  #pragma unroll
  for (int r = 0; r < 8; ++r) {
    invl[r] = 1.0f / lrun[r];
    grow[r] = gate[rowbase + r + 8 * half];
  }
  if (pass == 0) {
    #pragma unroll
    for (int i = 0; i < 16; ++i)
      #pragma unroll
      for (int r = 0; r < 8; ++r) {
        const int n = rowbase + r + 8 * half;
        const int c = i * 16 + l15;
        memfg[(size_t)n * CDIM + c] = acc[i][r] * invl[r] * grow[r];
      }
  } else {
    #pragma unroll
    for (int i = 0; i < 16; ++i)
      #pragma unroll
      for (int r = 0; r < 8; ++r) {
        const int n = rowbase + r + 8 * half;
        const int c = i * 16 + l15;
        float xv = (float)xn16[(size_t)n * CDIM + c];
        float f  = xv + memfg[(size_t)n * CDIM + c] - acc[i][r] * invl[r] * grow[r];
        const int bidx = n >> 12, hw = n & 4095;
        out[(((size_t)(bidx * CDIM + c)) << 12) + hw] = f;
      }
  }
}

// ---------------------------------------------------------------------------
extern "C" void kernel_launch(void* const* d_in, const int* in_sizes, int n_in,
                              void* d_out, int out_size, void* d_ws, size_t ws_size,
                              hipStream_t stream) {
  const float* feats = (const float*)d_in[0];
  const float* fgm   = (const float*)d_in[1];
  const float* bgm   = (const float*)d_in[2];
  const float* w1f   = (const float*)d_in[3];
  const float* b1f   = (const float*)d_in[4];
  const float* w2f   = (const float*)d_in[5];
  const float* b2f   = (const float*)d_in[6];
  const float* w1b   = (const float*)d_in[7];
  const float* b1b   = (const float*)d_in[8];
  const float* w2b   = (const float*)d_in[9];
  const float* b2b   = (const float*)d_in[10];
  float* out = (float*)d_out;

  // workspace layout (all offsets 256B aligned); total ~52.7 MB
  char* w = (char*)d_ws;
  h16*   xn16  = (h16*)(w);                                  // 16 MiB
  h16*   fgn16 = (h16*)(w + 16777216);                       // 1 MiB
  h16*   bgn16 = (h16*)(w + 16777216 + 1048576);             // 1 MiB
  float* gfg   = (float*)(w + 16777216 + 2 * 1048576);       // 128 KiB
  float* gbg   = (float*)(w + 16777216 + 2 * 1048576 + 131072);
  float* memfg = (float*)(w + 16777216 + 2 * 1048576 + 2 * 131072); // 32 MiB

  tm_prep_mem<<<dim3(MPAD, 2), 256, 0, stream>>>(fgm, bgm, fgn16, bgn16);
  tm_prep_x<<<NPIX / 32, 256, 0, stream>>>(feats, w1f, b1f, w2f, b2f,
                                           w1b, b1b, w2b, b2b,
                                           xn16, gfg, gbg);
  tm_attn<<<NPIX / 128, 256, 0, stream>>>(xn16, fgn16, gfg, memfg, out, 0);
  tm_attn<<<NPIX / 128, 256, 0, stream>>>(xn16, bgn16, gbg, memfg, out, 1);
}
